// Classifier_heterogeneous_76209899701046
// MI455X (gfx1250) — matrix operand order repaired and checked by reading, not compiled, not run on hardware
//
#include <hip/hip_runtime.h>
#include <hip/hip_bf16.h>

#define IN_CH      128
#define EDGE_DIM    64
#define DEC_CH     256
#define K_TOT      320            // 2*IN_CH + EDGE_DIM
#define LDS_STRIDE 328            // 320 + 8 pad (ushorts); 328*2 = 656 B, 16B-aligned rows

typedef __attribute__((ext_vector_type(16))) __bf16 v16bf;
typedef __attribute__((ext_vector_type(8)))  __bf16 v8bf;
typedef __attribute__((ext_vector_type(8)))  float  v8f;

#define CAT16(a, b) __builtin_shufflevector(a, b, 0,1,2,3,4,5,6,7,8,9,10,11,12,13,14,15)

static __device__ __forceinline__ unsigned short f32_to_bf16(float f) {
  unsigned int u = __float_as_uint(f);
  u += 0x7FFFu + ((u >> 16) & 1u);          // round-to-nearest-even
  return (unsigned short)(u >> 16);
}
static __device__ __forceinline__ float bf16_to_f32(unsigned short h) {
  return __uint_as_float(((unsigned int)h) << 16);
}

// ---------------------------------------------------------------------------
// One-time: W1 [K=320][N=256] f32  ->  transposed bf16 hi/lo  [N=256][K=320]
// ---------------------------------------------------------------------------
__global__ void w1_split_kernel(const float* __restrict__ W1,
                                unsigned short* __restrict__ hi,
                                unsigned short* __restrict__ lo) {
  const int i = blockIdx.x * 256 + threadIdx.x;
  if (i >= K_TOT * DEC_CH) return;
  const int k = i / DEC_CH;
  const int n = i % DEC_CH;
  const float f = W1[i];
  const unsigned short h = f32_to_bf16(f);
  hi[n * K_TOT + k] = h;
  lo[n * K_TOT + k] = f32_to_bf16(f - bf16_to_f32(h));
}

// ---------------------------------------------------------------------------
// Main fused kernel: 16 edges per workgroup, 4 waves, WMMA bf16 split-precision
// ---------------------------------------------------------------------------
__global__ __launch_bounds__(128)
void edge_mlp_kernel(const float* __restrict__ xs,
                     const float* __restrict__ xi,
                     const long long* __restrict__ eidx,   // [2][E] int64
                     const float* __restrict__ efeat,
                     const float* __restrict__ offs,
                     const unsigned short* __restrict__ w1t_hi,
                     const unsigned short* __restrict__ w1t_lo,
                     const float* __restrict__ b1,
                     const float* __restrict__ w2,
                     const float* __restrict__ b2,
                     float* __restrict__ out,
                     int nEdges) {
  __shared__ unsigned short sAhi[16 * LDS_STRIDE];
  __shared__ unsigned short sAlo[16 * LDS_STRIDE];
  __shared__ float sPart[4][16];

  const int tid  = threadIdx.x;
  const int base = blockIdx.x * 16;

  // ---- Stage 1: gather 16 edges' concat features, split to bf16 hi/lo in LDS
  {
    const int e = tid >> 3;                 // edge within tile (8 threads/edge)
    const int j = tid & 7;
    int edge = base + e;
    if (edge >= nEdges) edge = nEdges - 1;  // clamp (store is guarded later)
    const long long src = eidx[edge];
    const long long dst = eidx[nEdges + edge];
    const float* rs = xs    + (size_t)src  * IN_CH;
    const float* re = efeat + (size_t)edge * EDGE_DIM;
    const float* ri = xi    + (size_t)dst  * IN_CH;
    #pragma unroll
    for (int it = 0; it < 10; ++it) {
      const int k = it * 32 + j * 4;        // never crosses the 128/192 seams
      const float* p = (k < IN_CH)            ? (rs + k)
                     : (k < IN_CH + EDGE_DIM) ? (re + (k - IN_CH))
                                              : (ri + (k - IN_CH - EDGE_DIM));
      const float4 v = *reinterpret_cast<const float4*>(p);
      const float f[4] = {v.x, v.y, v.z, v.w};
      union { unsigned short u[4]; uint2 q; } ph, pl;
      #pragma unroll
      for (int c = 0; c < 4; ++c) {
        const unsigned short h = f32_to_bf16(f[c]);
        ph.u[c] = h;
        pl.u[c] = f32_to_bf16(f[c] - bf16_to_f32(h));
      }
      *reinterpret_cast<uint2*>(&sAhi[e * LDS_STRIDE + k]) = ph.q;
      *reinterpret_cast<uint2*>(&sAlo[e * LDS_STRIDE + k]) = pl.q;
    }
  }
  __syncthreads();

  // ---- Stage 2: per-wave WMMA over 4 N-tiles, fused ELU + h@W2
  const int lane = tid & 31;
  const int wave = tid >> 5;
  const int col  = lane & 15;     // N within tile (B/C/D layout)
  const int half = lane >> 4;     // lane-half selects K sub-block

  float outAcc[8];
  #pragma unroll
  for (int r = 0; r < 8; ++r) outAcc[r] = 0.f;

  // A layout (16-bit, 16x32): lanes 0-15 M=0-15 K={V*2+h | V<4} and +16 for V>=4;
  // lanes 16-31 same rows, K offset +8.
  const int aBase = col * LDS_STRIDE + half * 8;

  for (int t = 0; t < 4; ++t) {
    const int N = (wave * 4 + t) * 16 + col;
    const float b1v = b1[N];
    const float w2v = w2[N];
    v8f acc;
    #pragma unroll
    for (int r = 0; r < 8; ++r) acc[r] = b1v;   // C init = bias (N lives in lane)

    // B layout (16-bit, 32x16): lane col = N; same K order as A (lanes 16-31 K offset +8)
    const unsigned short* bh = w1t_hi + (size_t)N * K_TOT + half * 8;
    const unsigned short* bl = w1t_lo + (size_t)N * K_TOT + half * 8;

    #pragma unroll
    for (int kt = 0; kt < 10; ++kt) {
      const int ka = aBase + kt * 32;
      const v8bf a0h = *reinterpret_cast<const v8bf*>(&sAhi[ka]);
      const v8bf a1h = *reinterpret_cast<const v8bf*>(&sAhi[ka + 16]);
      const v8bf a0l = *reinterpret_cast<const v8bf*>(&sAlo[ka]);
      const v8bf a1l = *reinterpret_cast<const v8bf*>(&sAlo[ka + 16]);
      const v8bf b0h = *reinterpret_cast<const v8bf*>(bh + kt * 32);
      const v8bf b1h = *reinterpret_cast<const v8bf*>(bh + kt * 32 + 16);
      const v8bf b0l = *reinterpret_cast<const v8bf*>(bl + kt * 32);
      const v8bf b1l = *reinterpret_cast<const v8bf*>(bl + kt * 32 + 16);
      const v16bf Ah = CAT16(a0h, a1h);
      const v16bf Al = CAT16(a0l, a1l);
      const v16bf Bh = CAT16(b0h, b1h);
      const v16bf Bl = CAT16(b0l, b1l);
      // Split-precision fp32-accurate product: Ah*Bh + Al*Bh + Ah*Bl
      acc = __builtin_amdgcn_wmma_f32_16x16x32_bf16(false, Ah, false, Bh,
                                                    (short)0, acc, false, false);
      acc = __builtin_amdgcn_wmma_f32_16x16x32_bf16(false, Al, false, Bh,
                                                    (short)0, acc, false, false);
      acc = __builtin_amdgcn_wmma_f32_16x16x32_bf16(false, Ah, false, Bl,
                                                    (short)0, acc, false, false);
    }

    // ELU + fold into h @ W2 (column N contribution)
    #pragma unroll
    for (int r = 0; r < 8; ++r) {
      const float x = acc[r];
      const float h = x > 0.f ? x : (__expf(x) - 1.f);
      outAcc[r] = fmaf(h, w2v, outAcc[r]);
    }
  }

  // ---- Stage 3: reduce columns (16 lanes of a half) then waves
  #pragma unroll
  for (int r = 0; r < 8; ++r) {
    float v = outAcc[r];
    v += __shfl_xor(v, 1, 32);
    v += __shfl_xor(v, 2, 32);
    v += __shfl_xor(v, 4, 32);
    v += __shfl_xor(v, 8, 32);
    outAcc[r] = v;
  }
  if (col == 0) {
    #pragma unroll
    for (int r = 0; r < 8; ++r) sPart[wave][half * 8 + r] = outAcc[r];
  }
  __syncthreads();

  if (tid < 16) {
    const int edge = base + tid;
    if (edge < nEdges) {
      const long long dst = eidx[nEdges + edge];
      out[edge] = sPart[0][tid] + sPart[1][tid] + sPart[2][tid] + sPart[3][tid]
                + b2[0] + offs[dst];
    }
  }
}

// ---------------------------------------------------------------------------
extern "C" void kernel_launch(void* const* d_in, const int* in_sizes, int n_in,
                              void* d_out, int out_size, void* d_ws, size_t ws_size,
                              hipStream_t stream) {
  const float*     xs    = (const float*)d_in[0];      // x_student  [N,128] f32
  const float*     xi    = (const float*)d_in[1];      // x_item     [N,128] f32
  const long long* eidx  = (const long long*)d_in[2];  // edge_label_index [2,E] i64
  const float*     efeat = (const float*)d_in[3];      // edge_feat  [E,64] f32
  const float*     offs  = (const float*)d_in[4];      // offset     [N,1]  f32
  const float*     W1    = (const float*)d_in[5];      // [320,256] f32
  const float*     b1    = (const float*)d_in[6];      // [256] f32
  const float*     W2    = (const float*)d_in[7];      // [256,1] f32
  const float*     b2    = (const float*)d_in[8];      // [1] f32
  float* out = (float*)d_out;

  const int nEdges = in_sizes[2] / 2;

  unsigned short* w1t_hi = (unsigned short*)d_ws;                 // 163,840 B
  unsigned short* w1t_lo = w1t_hi + (size_t)K_TOT * DEC_CH;       // 163,840 B
  (void)ws_size; (void)n_in; (void)out_size;

  w1_split_kernel<<<(K_TOT * DEC_CH + 255) / 256, 256, 0, stream>>>(W1, w1t_hi, w1t_lo);

  const int nBlocks = (nEdges + 15) / 16;
  edge_mlp_kernel<<<nBlocks, 128, 0, stream>>>(xs, xi, eidx, efeat, offs,
                                               w1t_hi, w1t_lo, b1, W2, b2,
                                               out, nEdges);
}
